// Attention_64338610094336
// MI455X (gfx1250) — matrix operand order repaired and checked by reading, not compiled, not run on hardware
//
#include <hip/hip_runtime.h>
#include <hip/hip_bf16.h>
#include <math.h>

// Sigmoid attention for MI455X (gfx1250, wave32, WMMA bf16 path).
// Q,K,V,Out all (B=8, D=128, N=2048) fp32, layout [b][d][n] (n fastest).
//
// Out[b,v,m] = sum_n V[b,v,n] * sigmoid( (1/sqrt(128)) * sum_d Q[b,d,n]*K[b,d,m] )
//
// One WG = 8 wave32s owns a 128v x 64m output tile. Per 32-row n-chunk:
//   stage Qt[n][d] (bf16) in LDS via coalesced b128 loads,
//   wave w computes S-tile (n-sub = w>>2, m-sub = w&3) with resident K frags
//   (pre-scaled by 1/sqrt(128)), sigmoid via v_exp/v_rcp -> bf16 scores in
//   LDS, then each wave accumulates its 16v x 64m strip with 4 WMMAs
//   (V frags = 4x global_load_b128).

typedef __attribute__((ext_vector_type(16))) __bf16 v16bf;
typedef __attribute__((ext_vector_type(8)))  __bf16 v8bf;
typedef __attribute__((ext_vector_type(2)))  __bf16 v2bf;
typedef __attribute__((ext_vector_type(8)))  float  v8f;
typedef __attribute__((ext_vector_type(4)))  float  v4f;

#define BATCH  8
#define DDIM   128
#define NSEQ   2048
#define MTILE  64              // m columns per workgroup
#define NCHUNK 32              // n rows per outer iteration (= WMMA K)
#define QPITCH (DDIM + 8)      // Qt row pitch in bf16 (272B, 16B aligned)
#define SPITCH (NCHUNK + 8)    // score row pitch in bf16 (80B, 16B aligned)

// sigmoid(x) = rcp(1 + exp2(x * -log2(e))), all hardware-rate ops.
__device__ __forceinline__ float fast_sigmoid(float x) {
    float e = __builtin_amdgcn_exp2f(x * -1.442695040888963f);
    return __builtin_amdgcn_rcpf(1.0f + e);
}

__global__ __launch_bounds__(256)
void sigmoid_attn_kernel(const float* __restrict__ Q,
                         const float* __restrict__ K,
                         const float* __restrict__ V,
                         float* __restrict__ Out) {
    // Qt: [n_local][d] bf16 (transposed Q chunk)   32 x 136 x 2 = 8704 B
    // Sc: [m_local][n_local] bf16 (sigmoid scores) 64 x 40  x 2 = 5120 B
    __shared__ __bf16 s_q[NCHUNK][QPITCH];
    __shared__ __bf16 s_sc[MTILE][SPITCH];

    const int tid  = threadIdx.x;
    const int lane = tid & 31;
    const int wave = tid >> 5;          // 0..7
    const int half = lane >> 4;         // 0 or 1
    const int lm   = lane & 15;         // 0..15
    const int msub = wave & 3;          // this wave's S-tile m sub-block (0..3)
    const int nsub = wave >> 2;         // this wave's S-tile n sub-block (0..1)

    const int m0 = blockIdx.x * MTILE;
    const int b  = blockIdx.y;
    const size_t bOff = (size_t)b * DDIM * NSEQ;

    const float scale = 0.08838834764831845f;   // 1/sqrt(128)

    // ---- Resident B operand of GEMM1: K[d, m0+16*msub+lm] * scale,
    // 4 slices of K=32. B layout: slice s, element i -> d = 32*s + 8*half + i (+8 for i>=8).
    const int mcol = m0 + 16 * msub + lm;
    v16bf bK[4];
#pragma unroll
    for (int s = 0; s < 4; ++s) {
#pragma unroll
        for (int i = 0; i < 16; ++i) {
            int d = 32 * s + 8 * half + i + (i >= 8 ? 8 : 0);
            bK[s][i] = (__bf16)(K[bOff + (size_t)d * NSEQ + mcol] * scale);
        }
    }

    // ---- Output accumulators: wave rows v = 16*wave..+15, 4 m sub-tiles.
    v8f acc[4];
#pragma unroll
    for (int t = 0; t < 4; ++t)
        acc[t] = (v8f){0.f,0.f,0.f,0.f,0.f,0.f,0.f,0.f};

    const float* vbase = &V[bOff + (size_t)(16 * wave + lm) * NSEQ];

    for (int n0 = 0; n0 < NSEQ; n0 += NCHUNK) {
        // ========== Stage Qt[n][d] = bf16(Q[d][n0+n]), coalesced ==========
        // 512 (d-pair, n4) units; 2 per thread. Unit u: p=u>>3 (d=2p,2p+1),
        // c=u&7 (n=4c..4c+3). Two b128 loads -> 4 packed b32 LDS stores.
#pragma unroll
        for (int r = 0; r < 2; ++r) {
            int u = tid + 256 * r;
            int p = u >> 3, c = u & 7;
            v4f f0 = *reinterpret_cast<const v4f*>(&Q[bOff + (size_t)(2 * p)     * NSEQ + n0 + 4 * c]);
            v4f f1 = *reinterpret_cast<const v4f*>(&Q[bOff + (size_t)(2 * p + 1) * NSEQ + n0 + 4 * c]);
#pragma unroll
            for (int j = 0; j < 4; ++j) {
                v2bf pk = {(__bf16)f0[j], (__bf16)f1[j]};
                *reinterpret_cast<v2bf*>(&s_q[4 * c + j][2 * p]) = pk;
            }
        }
        __syncthreads();

        // ========== Phase A: S-tile (16 n rows, 16 m cols) ==========
        // A layout per lane: elements 0..7  -> d = 32s + 8*half + i,
        //                    elements 8..15 -> d = 32s + 16 + 8*half + (i-8).
        const __bf16* qrow = &s_q[16 * nsub + lm][0];
        v8f x = (v8f){0.f,0.f,0.f,0.f,0.f,0.f,0.f,0.f};
#pragma unroll
        for (int s = 0; s < 4; ++s) {
            v8bf lo = *reinterpret_cast<const v8bf*>(qrow + 32 * s + 8 * half);
            v8bf hi = *reinterpret_cast<const v8bf*>(qrow + 32 * s + 16 + 8 * half);
            v16bf aQ;
#pragma unroll
            for (int i = 0; i < 8; ++i) { aQ[i] = lo[i]; aQ[i + 8] = hi[i]; }
            x = __builtin_amdgcn_wmma_f32_16x16x32_bf16(
                    false, aQ, false, bK[s], (short)0, x, false, false);
        }
        // sigmoid + packed bf16 pair stores: C element r -> n = 16*nsub + r + 8*half
        {
            __bf16* srow = &s_sc[16 * msub + lm][16 * nsub + 8 * half];
#pragma unroll
            for (int rp = 0; rp < 4; ++rp) {
                float s0 = fast_sigmoid(x[2 * rp]);
                float s1 = fast_sigmoid(x[2 * rp + 1]);
                v2bf pk = {(__bf16)s0, (__bf16)s1};
                *reinterpret_cast<v2bf*>(srow + 2 * rp) = pk;
            }
        }
        __syncthreads();

        // ========== Phase B: acc[mt] += V[16*wave.., n-chunk] x Sc ==========
        // A operand (V): per lane two contiguous 8-float runs.
        v16bf aV;
        {
            v4f r0 = *reinterpret_cast<const v4f*>(vbase + n0 + 8 * half);
            v4f r1 = *reinterpret_cast<const v4f*>(vbase + n0 + 8 * half + 4);
            v4f r2 = *reinterpret_cast<const v4f*>(vbase + n0 + 16 + 8 * half);
            v4f r3 = *reinterpret_cast<const v4f*>(vbase + n0 + 16 + 8 * half + 4);
#pragma unroll
            for (int i = 0; i < 4; ++i) {
                aV[i]      = (__bf16)r0[i];
                aV[i + 4]  = (__bf16)r1[i];
                aV[i + 8]  = (__bf16)r2[i];
                aV[i + 12] = (__bf16)r3[i];
            }
        }
#pragma unroll
        for (int mt = 0; mt < 4; ++mt) {
            // B operand (scores): element i -> n_local = 8*half + i (+8 for i>=8), col = lm.
            const __bf16* srow = &s_sc[16 * mt + lm][8 * half];
            v8bf lo = *reinterpret_cast<const v8bf*>(srow);
            v8bf hi = *reinterpret_cast<const v8bf*>(srow + 16);
            v16bf bS;
#pragma unroll
            for (int i = 0; i < 8; ++i) { bS[i] = lo[i]; bS[i + 8] = hi[i]; }
            acc[mt] = __builtin_amdgcn_wmma_f32_16x16x32_bf16(
                          false, aV, false, bS, (short)0, acc[mt], false, false);
        }
        __syncthreads();
    }

    // ---- Store output: row v = 16*wave + r + 8*half, col m0 + 16*mt + lm.
#pragma unroll
    for (int mt = 0; mt < 4; ++mt) {
#pragma unroll
        for (int r = 0; r < 8; ++r) {
            int vr = 16 * wave + r + 8 * half;
            Out[bOff + (size_t)vr * NSEQ + (m0 + 16 * mt + lm)] = acc[mt][r];
        }
    }
}

extern "C" void kernel_launch(void* const* d_in, const int* in_sizes, int n_in,
                              void* d_out, int out_size, void* d_ws, size_t ws_size,
                              hipStream_t stream) {
    const float* Q = (const float*)d_in[0];
    const float* K = (const float*)d_in[1];
    const float* V = (const float*)d_in[2];
    float* Out     = (float*)d_out;

    dim3 grid(NSEQ / MTILE, BATCH);   // 32 m-tiles x 8 batches = 256 WGs
    sigmoid_attn_kernel<<<grid, 256, 0, stream>>>(Q, K, V, Out);
}
